// Head_55155970015649
// MI455X (gfx1250) — matrix operand order repaired and checked by reading, not compiled, not run on hardware
//
#include <hip/hip_runtime.h>

#define HEAD  64
#define NEMBD 384
#define BATCH 8
#define SEQ   4096

typedef __attribute__((ext_vector_type(16))) __bf16 v16bf;
typedef __attribute__((ext_vector_type(8)))  __bf16 v8bf;
typedef __attribute__((ext_vector_type(8)))  float  v8f;

// LDS chunk layout (bf16 elements), per buffer: exactly 8192 elements (16 KB)
#define KSTRIDE 88            // 32 key rows x 64 d (+pad): 176B rows, conflict-free
#define VSTRIDE 40            // 64 d rows x 32 keys (+pad): 80B rows, conflict-free
#define K2OFF   2816          // 32*88
#define VTOFF   5632          // 2*32*88
#define BUFELEMS 8192         // 5632 + 64*40

static __device__ __forceinline__ v16bf join8(v8bf a, v8bf b) {
  v16bf r;
#pragma unroll
  for (int i = 0; i < 8; ++i) { r[i] = a[i]; r[i + 8] = b[i]; }
  return r;
}

// B-operand: 16 contiguous bf16 (lane = column N, K contiguous per half-wave)
static __device__ __forceinline__ v16bf loadB16(const __bf16* p) {
  return join8(*(const v8bf*)p, *(const v8bf*)(p + 8));
}

// A-operand: 16 contiguous K -> p[0..7] and p[8..15]
static __device__ __forceinline__ v16bf loadA16(const __bf16* p) {
  return join8(*(const v8bf*)p, *(const v8bf*)(p + 8));
}

// A-operand built from f32 source with on-the-fly bf16 conversion
static __device__ __forceinline__ v16bf cvtA_f32(const float* p) {
  v16bf r;
#pragma unroll
  for (int i = 0; i < 8; ++i) {
    r[i]     = (__bf16)p[i];
    r[i + 8] = (__bf16)p[8 + i];
  }
  return r;
}

static __device__ __forceinline__ v8f wmma_bf16(v16bf a, v16bf b, v8f c) {
  return __builtin_amdgcn_wmma_f32_16x16x32_bf16(false, a, false, b,
                                                 (short)0, c, false, false);
}

// CDNA5 async DMA: 16B global -> LDS, tracked by ASYNCcnt (no VGPR round-trip)
static __device__ __forceinline__ void async_cp16(unsigned ldsByte,
                                                  const __bf16* g) {
  asm volatile("global_load_async_to_lds_b128 %0, %1, off"
               :: "v"(ldsByte), "v"(g) : "memory");
}
static __device__ __forceinline__ void wait_async0() {
  asm volatile("s_wait_asynccnt 0x0" ::: "memory");
}

// ---------------------------------------------------------------------------
// Kernel 1: transpose W -> WT[5][64][384] bf16; fold 1/sqrt(64) into Wq1, Wq2
// ---------------------------------------------------------------------------
__global__ void wt_kernel(const float* __restrict__ Wq1, const float* __restrict__ Wk1,
                          const float* __restrict__ Wq2, const float* __restrict__ Wk2,
                          const float* __restrict__ Wv, __bf16* __restrict__ wt) {
  int idx = blockIdx.x * 256 + threadIdx.x;
  if (idx >= 5 * HEAD * NEMBD) return;
  int m   = idx / (HEAD * NEMBD);
  int rem = idx % (HEAD * NEMBD);
  int h = rem / NEMBD, c = rem % NEMBD;
  const float* W = (m == 0) ? Wq1 : (m == 1) ? Wk1 : (m == 2) ? Wq2
                 : (m == 3) ? Wk2 : Wv;
  float s = (m == 0 || m == 2) ? 0.125f : 1.0f;  // exact in bf16
  wt[idx] = (__bf16)(W[c * HEAD + h] * s);
}

// ---------------------------------------------------------------------------
// Kernel 2: projections. One wave per 16-row tile of X[32768][384].
// B operand double-buffered so loads overlap WMMA (wait loadcnt<=1, not 0).
// ---------------------------------------------------------------------------
__global__ void __launch_bounds__(128)
proj_kernel(const float* __restrict__ x, const __bf16* __restrict__ wt,
            __bf16* __restrict__ q1, __bf16* __restrict__ k1,
            __bf16* __restrict__ q2, __bf16* __restrict__ k2,
            __bf16* __restrict__ vt) {
  const int lane  = threadIdx.x & 31;
  const int wave  = threadIdx.x >> 5;
  const int g     = blockIdx.x * 4 + wave;     // 0..2047 row tiles
  const int r0    = g * 16;
  const int nL    = lane & 15;
  const int hi    = lane >> 4;
  const int khalf = hi * 16;

  // Load X tile: 12 A-registers covering K=384
  v16bf a[12];
  const float* xr = x + (size_t)(r0 + nL) * NEMBD + khalf;
#pragma unroll
  for (int kk = 0; kk < 12; ++kk) a[kk] = cvtA_f32(xr + kk * 32);

  __bf16* const outs[4] = {q1, k1, q2, k2};

#pragma unroll
  for (int im = 0; im < 5; ++im) {
    const __bf16* wrow = wt + im * HEAD * NEMBD + hi * 16;
#pragma unroll
    for (int nt = 0; nt < 4; ++nt) {
      v8f acc = {0.f, 0.f, 0.f, 0.f, 0.f, 0.f, 0.f, 0.f};
      const __bf16* wp = wrow + (size_t)(nt * 16 + nL) * NEMBD;
      v16bf bcur = loadB16(wp);
#pragma unroll
      for (int kk = 0; kk < 12; ++kk) {
        v16bf bnext = (kk < 11) ? loadB16(wp + (kk + 1) * 32) : bcur;
        acc = wmma_bf16(a[kk], bcur, acc);
        bcur = bnext;
      }

      if (im < 4) {
        __bf16* o = outs[im];
#pragma unroll
        for (int r = 0; r < 8; ++r)
          o[(size_t)(r0 + r + hi * 8) * HEAD + nt * 16 + nL] = (__bf16)acc[r];
      } else {
        // V transposed: vt[b][d][t]; lane's 8 values are contiguous in t
        const int b  = r0 >> 12;
        const int t0 = r0 & (SEQ - 1);
        const int d  = nt * 16 + nL;
        v8bf pk;
#pragma unroll
        for (int r = 0; r < 8; ++r) pk[r] = (__bf16)acc[r];
        *(v8bf*)(vt + ((size_t)b * HEAD + d) * SEQ + t0 + hi * 8) = pk;
      }
    }
  }
}

// ---------------------------------------------------------------------------
// Kernel 3: dual flash attention. 4 waves/block share the key stream of one
// batch: K1/K2/V^T chunks staged once via async-to-LDS, double-buffered.
// ---------------------------------------------------------------------------
__global__ void __launch_bounds__(128)
attn_kernel(const __bf16* __restrict__ q1, const __bf16* __restrict__ k1,
            const __bf16* __restrict__ q2, const __bf16* __restrict__ k2,
            const __bf16* __restrict__ vt, const float* __restrict__ lambp,
            float* __restrict__ out) {
  __shared__ __bf16 smem[2 * BUFELEMS];   // 32 KB

  const int tid   = threadIdx.x;
  const int lane  = tid & 31;
  const int wave  = tid >> 5;
  const int g     = blockIdx.x * 4 + wave;      // 0..2047
  const int b     = g >> 8;                     // all 4 waves: same batch
  const int t0    = (g & 255) << 4;
  const int nL    = lane & 15;
  const int hi    = lane >> 4;
  const int khalf = hi * 16;
  const float lamb = *lambp;

  const size_t bRow = (size_t)b * SEQ;
  const __bf16* const Qp[2] = {q1, q2};
  const __bf16* k1b = k1 + bRow * HEAD;
  const __bf16* k2b = k2 + bRow * HEAD;
  const __bf16* vtb = vt + (size_t)b * HEAD * SEQ;
  const unsigned smemBase = (unsigned)(size_t)(void*)smem;

  // Stage one 32-key chunk (K1:4KB, K2:4KB, V^T:4KB) = 768 x b128, 6/thread
  auto stage = [&](int chunk, int bufIdx) {
    const int s0 = chunk * 32;
    const unsigned bb = smemBase + (unsigned)bufIdx * (BUFELEMS * 2);
#pragma unroll
    for (int i = 0; i < 6; ++i) {
      const int o   = tid + i * 128;
      const int mat = o >> 8;
      const int rem = o & 255;
      const __bf16* gp;
      unsigned lds;
      if (mat < 2) {
        const int row = rem >> 3, seg = rem & 7;
        gp  = (mat ? k2b : k1b) + (size_t)(s0 + row) * HEAD + seg * 8;
        lds = bb + 2u * (unsigned)(mat * K2OFF + row * KSTRIDE + seg * 8);
      } else {
        const int row = rem >> 2, seg = rem & 3;   // row = d
        gp  = vtb + (size_t)row * SEQ + s0 + seg * 8;
        lds = bb + 2u * (unsigned)(VTOFF + row * VSTRIDE + seg * 8);
      }
      async_cp16(lds, gp);
    }
  };

  // Q B-operands: lane = query column, K(=head dim) contiguous
  v16bf bq[2][2];
#pragma unroll
  for (int mi = 0; mi < 2; ++mi) {
    const __bf16* qrow = Qp[mi] + (bRow + t0 + nL) * HEAD + hi * 16;
    bq[mi][0] = loadB16(qrow);
    bq[mi][1] = loadB16(qrow + 32);
  }

  v8f O[2][4];
#pragma unroll
  for (int mi = 0; mi < 2; ++mi)
#pragma unroll
    for (int dt = 0; dt < 4; ++dt)
      O[mi][dt] = (v8f){0.f, 0.f, 0.f, 0.f, 0.f, 0.f, 0.f, 0.f};
  float mrun[2] = {-INFINITY, -INFINITY};
  float lrun[2] = {0.f, 0.f};

  stage(0, 0);
  wait_async0();
  __syncthreads();

  for (int it = 0; it < SEQ / 32; ++it) {
    if (it + 1 < SEQ / 32) stage(it + 1, (it + 1) & 1);  // overlap DMA

    const __bf16* buf = smem + (it & 1) * BUFELEMS;
    const __bf16* kbuf[2] = {buf, buf + K2OFF};
    const __bf16* vbuf = buf + VTOFF;

    v16bf bp[2];
#pragma unroll
    for (int mi = 0; mi < 2; ++mi) {
      // S^T tiles from LDS: 2 key sub-tiles of 16, K=64 over 2 WMMA steps
      v8f st[2];
#pragma unroll
      for (int j = 0; j < 2; ++j) {
        const __bf16* krow = kbuf[mi] + (j * 16 + nL) * KSTRIDE + khalf;
        v8f acc = {0.f, 0.f, 0.f, 0.f, 0.f, 0.f, 0.f, 0.f};
        acc = wmma_bf16(loadA16(krow), bq[mi][0], acc);
        acc = wmma_bf16(loadA16(krow + 32), bq[mi][1], acc);
        st[j] = acc;
      }
      // online softmax over this 32-key chunk (per query = per lane pair)
      float cm = st[0][0];
#pragma unroll
      for (int r = 1; r < 8; ++r) cm = fmaxf(cm, st[0][r]);
#pragma unroll
      for (int r = 0; r < 8; ++r) cm = fmaxf(cm, st[1][r]);
      cm = fmaxf(cm, __shfl_xor(cm, 16));
      const float mn    = fmaxf(mrun[mi], cm);
      const float alpha = __expf(mrun[mi] - mn);
      float p0[8], p1[8], rs = 0.f;
#pragma unroll
      for (int r = 0; r < 8; ++r) { p0[r] = __expf(st[0][r] - mn); rs += p0[r]; }
#pragma unroll
      for (int r = 0; r < 8; ++r) { p1[r] = __expf(st[1][r] - mn); rs += p1[r]; }
      rs += __shfl_xor(rs, 16);
      lrun[mi] = lrun[mi] * alpha + rs;
      mrun[mi] = mn;
#pragma unroll
      for (int dt = 0; dt < 4; ++dt) O[mi][dt] *= alpha;
      // Build P^T B-operand: half-wave swap via shfl_xor(16)
      v16bf bpv;
#pragma unroll
      for (int r = 0; r < 8; ++r) {
        float send = hi ? p0[r] : p1[r];
        float recv = __shfl_xor(send, 16);
        bpv[r]     = (__bf16)(hi ? recv : p0[r]);
        bpv[r + 8] = (__bf16)(hi ? p1[r] : recv);
      }
      bp[mi] = bpv;
    }
    // out^T += V^T . P^T  (V^T A-operands from LDS, shared by both maps)
#pragma unroll
    for (int dt = 0; dt < 4; ++dt) {
      const __bf16* vrow = vbuf + (dt * 16 + nL) * VSTRIDE + khalf;
      v16bf av = loadA16(vrow);
      O[0][dt] = wmma_bf16(av, bp[0], O[0][dt]);
      O[1][dt] = wmma_bf16(av, bp[1], O[1][dt]);
    }

    wait_async0();       // next chunk's DMA done (this wave)
    __syncthreads();     // all waves' DMA visible before buffer swap
  }

  // Combine: out = O1/l1 - lamb*O2/l2 ; out^T C-layout -> contiguous f32 rows
  const float inv1 = 1.0f / lrun[0];
  const float inv2 = lamb / lrun[1];
#pragma unroll
  for (int dt = 0; dt < 4; ++dt) {
    float* orow = out + (bRow + t0 + nL) * HEAD + dt * 16 + hi * 8;
    float4 lo, hh;
    lo.x = O[0][dt][0] * inv1 - O[1][dt][0] * inv2;
    lo.y = O[0][dt][1] * inv1 - O[1][dt][1] * inv2;
    lo.z = O[0][dt][2] * inv1 - O[1][dt][2] * inv2;
    lo.w = O[0][dt][3] * inv1 - O[1][dt][3] * inv2;
    hh.x = O[0][dt][4] * inv1 - O[1][dt][4] * inv2;
    hh.y = O[0][dt][5] * inv1 - O[1][dt][5] * inv2;
    hh.z = O[0][dt][6] * inv1 - O[1][dt][6] * inv2;
    hh.w = O[0][dt][7] * inv1 - O[1][dt][7] * inv2;
    *(float4*)orow = lo;
    *(float4*)(orow + 4) = hh;
  }
}

// ---------------------------------------------------------------------------
extern "C" void kernel_launch(void* const* d_in, const int* in_sizes, int n_in,
                              void* d_out, int out_size, void* d_ws, size_t ws_size,
                              hipStream_t stream) {
  const float* x    = (const float*)d_in[0];
  const float* lamb = (const float*)d_in[1];
  const float* Wq1  = (const float*)d_in[2];
  const float* Wk1  = (const float*)d_in[3];
  const float* Wq2  = (const float*)d_in[4];
  const float* Wk2  = (const float*)d_in[5];
  const float* Wv   = (const float*)d_in[6];
  float* outp = (float*)d_out;

  // Workspace: WT (5*64*384) + Q1,K1,Q2,K2 (row-major) + V^T  — all bf16
  const size_t msz = (size_t)BATCH * SEQ * HEAD;
  __bf16* wt = (__bf16*)d_ws;
  __bf16* q1 = wt + 5 * HEAD * NEMBD;
  __bf16* k1 = q1 + msz;
  __bf16* q2 = k1 + msz;
  __bf16* k2 = q2 + msz;
  __bf16* vt = k2 + msz;

  wt_kernel<<<(5 * HEAD * NEMBD + 255) / 256, 256, 0, stream>>>(
      Wq1, Wk1, Wq2, Wk2, Wv, wt);
  proj_kernel<<<(BATCH * SEQ / 16) / 4, 128, 0, stream>>>(
      x, wt, q1, k1, q2, k2, vt);
  attn_kernel<<<(BATCH * SEQ / 16) / 4, 128, 0, stream>>>(
      q1, k1, q2, k2, vt, lamb, outp);
}
